// HierarchicalSelectiveSSMLayer_34969623724661
// MI455X (gfx1250) — compile-verified, matrix operand order checked by reading, not run on hardware
//
#include <hip/hip_runtime.h>
#include <math.h>

typedef __attribute__((ext_vector_type(16))) __bf16 v16bf;
typedef __attribute__((ext_vector_type(8)))  __bf16 v8bf;
typedef __attribute__((ext_vector_type(8)))  float  v8f;

#define D_MODEL 1024
#define TSEQ    4096
#define BATCH   4
#define MTOT    (BATCH*TSEQ)   // 16384 tokens
#define KDIM    1024
#define CHUNK   256
#define NCHUNK  (TSEQ/CHUNK)   // 16

union BFrag { v16bf v; v8bf h[2]; };

// Fragment load: two contiguous 16B chunks (K = base..base+7 and +16..+23),
// matching CDNA5 16-bit A/B VGPR layout (ISA 7.12.2) with lane>>4 K-half.
__device__ __forceinline__ void load_frag(BFrag& f, const __bf16* p)
{
  const v8bf* q = (const v8bf*)p;
  f.h[0] = q[0];
  f.h[1] = q[2];
}

// ---------------------------------------------------------------------------
// Core WMMA mainloop: one wave computes a 64(M) x 16(N) strip, K = 1024.
// A is M x K bf16 row-major, Bt is N x K bf16 row-major (pre-transposed).
// Two-stage software pipeline (ping/pong fragment buffers, K-unroll of 64)
// so fragment loads for step k+32 are in flight while WMMAs consume step k.
// ---------------------------------------------------------------------------
__device__ __forceinline__ void wmma_strip64x16(
    const __bf16* __restrict__ A, const __bf16* __restrict__ Bt,
    int m0, int n0, v8f acc[4])
{
  const int lane = threadIdx.x & 31;
  const int row  = lane & 15;
  const int half = lane >> 4;
  const __bf16* brow  = Bt + (size_t)(n0 + row) * KDIM + half * 8;
  const __bf16* arow0 = A  + (size_t)(m0 + row) * KDIM + half * 8;

  BFrag fb0, fb1, fa0[4], fa1[4];

  // prologue: stage 0 fragments for k = 0
  load_frag(fb0, brow);
  #pragma unroll
  for (int t = 0; t < 4; ++t) load_frag(fa0[t], arow0 + (size_t)t * 16 * KDIM);

  for (int k0 = 0; k0 < KDIM; k0 += 64) {
    // prefetch stage 1 (k0 + 32) while stage 0 computes
    load_frag(fb1, brow + k0 + 32);
    #pragma unroll
    for (int t = 0; t < 4; ++t)
      load_frag(fa1[t], arow0 + (size_t)t * 16 * KDIM + k0 + 32);

    #pragma unroll
    for (int t = 0; t < 4; ++t)
      acc[t] = __builtin_amdgcn_wmma_f32_16x16x32_bf16(
          false, fa0[t].v, false, fb0.v, (short)0, acc[t], false, false);

    // prefetch stage 0 (k0 + 64) while stage 1 computes
    if (k0 + 64 < KDIM) {
      load_frag(fb0, brow + k0 + 64);
      #pragma unroll
      for (int t = 0; t < 4; ++t)
        load_frag(fa0[t], arow0 + (size_t)t * 16 * KDIM + k0 + 64);
    }

    #pragma unroll
    for (int t = 0; t < 4; ++t)
      acc[t] = __builtin_amdgcn_wmma_f32_16x16x32_bf16(
          false, fa1[t].v, false, fb1.v, (short)0, acc[t], false, false);
  }
}

// ---------------------------------------------------------------------------
// Weight convert + transpose: W (K x N fp32, row-major) -> Wt (N x K bf16)
// ---------------------------------------------------------------------------
__global__ void cvt_tr_kernel(const float* __restrict__ W, __bf16* __restrict__ Wt,
                              int K, int N)
{
  size_t idx = (size_t)blockIdx.x * 256 + threadIdx.x;
  if (idx >= (size_t)K * N) return;
  int n = (int)(idx / K);
  int k = (int)(idx % K);
  Wt[idx] = (__bf16)W[(size_t)k * N + n];
}

// ---------------------------------------------------------------------------
// LayerNorm: one block per token, fp32 in -> bf16 out
// ---------------------------------------------------------------------------
__global__ void ln_kernel(const float* __restrict__ x, const float* __restrict__ g,
                          const float* __restrict__ b, __bf16* __restrict__ xn)
{
  __shared__ float s1[256], s2[256];
  const int tok = blockIdx.x;
  const float* xr = x + (size_t)tok * D_MODEL;
  float sum = 0.f, sq = 0.f;
  for (int i = threadIdx.x; i < D_MODEL; i += 256) {
    float v = xr[i]; sum += v; sq += v * v;
  }
  s1[threadIdx.x] = sum; s2[threadIdx.x] = sq;
  __syncthreads();
  for (int off = 128; off > 0; off >>= 1) {
    if (threadIdx.x < off) {
      s1[threadIdx.x] += s1[threadIdx.x + off];
      s2[threadIdx.x] += s2[threadIdx.x + off];
    }
    __syncthreads();
  }
  float mu  = s1[0] * (1.0f / D_MODEL);
  float var = s2[0] * (1.0f / D_MODEL) - mu * mu;
  float rs  = rsqrtf(var + 1e-5f);
  __bf16* o = xn + (size_t)tok * D_MODEL;
  for (int i = threadIdx.x; i < D_MODEL; i += 256)
    o[i] = (__bf16)((xr[i] - mu) * rs * g[i] + b[i]);
}

// ---------------------------------------------------------------------------
// GEMM 1: xn @ W_in + b_in -> xz fp32 (M x 2048)
// ---------------------------------------------------------------------------
__global__ void __launch_bounds__(128)
gemm_in_kernel(const __bf16* __restrict__ A, const __bf16* __restrict__ Bt,
               const float* __restrict__ bias, float* __restrict__ xz)
{
  const int wave = threadIdx.x >> 5;
  const int m0 = blockIdx.x * 64;
  const int n0 = blockIdx.y * 64 + wave * 16;
  v8f acc[4] = {};
  wmma_strip64x16(A, Bt, m0, n0, acc);
  const int lane = threadIdx.x & 31;
  const int n    = n0 + (lane & 15);
  const int half = lane >> 4;
  const float bn = bias[n];
  #pragma unroll
  for (int t = 0; t < 4; ++t)
    #pragma unroll
    for (int v = 0; v < 8; ++v) {
      int m = m0 + t * 16 + v + 8 * half;
      xz[(size_t)m * (2 * D_MODEL) + n] = acc[t][v] + bn;
    }
}

// ---------------------------------------------------------------------------
// Depthwise conv k=5, pad 2: x_proj (= xz[:, :D]) -> x_conv bf16
// ---------------------------------------------------------------------------
__global__ void conv_kernel(const float* __restrict__ xz, const float* __restrict__ ck,
                            const float* __restrict__ cb, __bf16* __restrict__ xconv)
{
  size_t idx = (size_t)blockIdx.x * 256 + threadIdx.x;   // over MTOT * D
  int d = (int)(idx & (D_MODEL - 1));
  int m = (int)(idx >> 10);
  int t = m & (TSEQ - 1);
  float acc = cb[d];
  #pragma unroll
  for (int i = 0; i < 5; ++i) {
    int tt = t + i - 2;
    if (tt >= 0 && tt < TSEQ)
      acc += xz[(size_t)(m + (tt - t)) * (2 * D_MODEL) + d] * ck[i * D_MODEL + d];
  }
  xconv[idx] = (__bf16)acc;
}

// ---------------------------------------------------------------------------
// Gate GEMM: x_conv @ [Wa|Wb|Wc|Wd] with fused activation epilogue.
// Column segment 0: tanh(a0+v+ba), 1: sigmoid(b0+v+bb), 2: sigmoid(c0+v+bc),
// 3: v+bd+d0.  Output bf16 (M x 4096).
// ---------------------------------------------------------------------------
__global__ void __launch_bounds__(128)
gemm_gates_kernel(const __bf16* __restrict__ A, const __bf16* __restrict__ Bt,
                  const float* __restrict__ ba, const float* __restrict__ bb,
                  const float* __restrict__ bc, const float* __restrict__ bd,
                  const float* __restrict__ a0, const float* __restrict__ b0,
                  const float* __restrict__ c0, const float* __restrict__ d0,
                  __bf16* __restrict__ gates)
{
  const int wave = threadIdx.x >> 5;
  const int m0 = blockIdx.x * 64;
  const int n0 = blockIdx.y * 64 + wave * 16;
  v8f acc[4] = {};
  wmma_strip64x16(A, Bt, m0, n0, acc);
  const int lane = threadIdx.x & 31;
  const int n    = n0 + (lane & 15);
  const int half = lane >> 4;
  const int seg  = n >> 10;
  const int j    = n & (D_MODEL - 1);
  float base, pb;
  if      (seg == 0) { base = a0[j]; pb = ba[j]; }
  else if (seg == 1) { base = b0[j]; pb = bb[j]; }
  else if (seg == 2) { base = c0[j]; pb = bc[j]; }
  else               { base = d0[j]; pb = bd[j]; }
  #pragma unroll
  for (int t = 0; t < 4; ++t)
    #pragma unroll
    for (int v = 0; v < 8; ++v) {
      int m = m0 + t * 16 + v + 8 * half;
      float val = acc[t][v] + pb + base;
      if (seg == 0)      val = tanhf(val);
      else if (seg <= 2) val = 1.f / (1.f + __expf(-val));
      gates[(size_t)m * 4096 + n] = (__bf16)val;
    }
}

// ---------------------------------------------------------------------------
// Chunked affine scan:  s_t = a_t * s_{t-1} + b_t * x_t
// pass1: per chunk compute (P = prod a, S = scan-from-zero)
// pass2: sequential compose over chunks -> chunk start states
// pass3: rescan chunk from start state, emit y = (c*s + d*x) * sigmoid(z)
// idx layout: idx = ((b * NCHUNK + c) << 10) | d
// ---------------------------------------------------------------------------
__global__ void scan1_kernel(const __bf16* __restrict__ gates,
                             const float* __restrict__ xz,
                             float* __restrict__ chP, float* __restrict__ chS)
{
  int idx = blockIdx.x * 256 + threadIdx.x;           // 65536 threads
  int d = idx & (D_MODEL - 1);
  int c = (idx >> 10) & (NCHUNK - 1);
  int b = idx >> 14;
  float P = 1.f, S = 0.f;
  size_t mbase = (size_t)b * TSEQ + (size_t)c * CHUNK;
  for (int t = 0; t < CHUNK; ++t) {
    size_t m = mbase + t;
    const __bf16* g = gates + m * 4096;
    float at = (float)g[d];
    float bt = (float)g[D_MODEL + d];
    float xt = xz[m * (2 * D_MODEL) + d];
    P *= at;
    S = at * S + bt * xt;
  }
  chP[idx] = P; chS[idx] = S;
}

__global__ void scan2_kernel(const float* __restrict__ chP,
                             const float* __restrict__ chS,
                             float* __restrict__ h0)
{
  int idx = blockIdx.x * 256 + threadIdx.x;           // 4096 threads
  int d = idx & (D_MODEL - 1);
  int b = idx >> 10;
  float h = 0.f;
  for (int c = 0; c < NCHUNK; ++c) {
    int o = ((b * NCHUNK + c) << 10) | d;
    h0[o] = h;
    h = chP[o] * h + chS[o];
  }
}

__global__ void scan3_kernel(const __bf16* __restrict__ gates,
                             const float* __restrict__ xz,
                             const float* __restrict__ h0,
                             __bf16* __restrict__ ybuf)
{
  int idx = blockIdx.x * 256 + threadIdx.x;           // 65536 threads
  int d = idx & (D_MODEL - 1);
  int c = (idx >> 10) & (NCHUNK - 1);
  int b = idx >> 14;
  float s = h0[idx];
  size_t mbase = (size_t)b * TSEQ + (size_t)c * CHUNK;
  for (int t = 0; t < CHUNK; ++t) {
    size_t m = mbase + t;
    const __bf16* g = gates + m * 4096;
    float at = (float)g[d];
    float bt = (float)g[D_MODEL + d];
    float ct = (float)g[2 * D_MODEL + d];
    float dt = (float)g[3 * D_MODEL + d];
    float xt = xz[m * (2 * D_MODEL) + d];
    float z  = xz[m * (2 * D_MODEL) + D_MODEL + d];
    s = at * s + bt * xt;
    float y = ct * s + dt * xt;
    y *= 1.f / (1.f + __expf(-z));
    ybuf[m * D_MODEL + d] = (__bf16)y;
  }
}

// ---------------------------------------------------------------------------
// Output GEMM: y @ W_out + b_out + x(residual) -> d_out fp32
// ---------------------------------------------------------------------------
__global__ void __launch_bounds__(128)
gemm_out_kernel(const __bf16* __restrict__ A, const __bf16* __restrict__ Bt,
                const float* __restrict__ bias, const float* __restrict__ xres,
                float* __restrict__ out)
{
  const int wave = threadIdx.x >> 5;
  const int m0 = blockIdx.x * 64;
  const int n0 = blockIdx.y * 64 + wave * 16;
  v8f acc[4] = {};
  wmma_strip64x16(A, Bt, m0, n0, acc);
  const int lane = threadIdx.x & 31;
  const int n    = n0 + (lane & 15);
  const int half = lane >> 4;
  const float bn = bias[n];
  #pragma unroll
  for (int t = 0; t < 4; ++t)
    #pragma unroll
    for (int v = 0; v < 8; ++v) {
      int m = m0 + t * 16 + v + 8 * half;
      out[(size_t)m * D_MODEL + n] = acc[t][v] + bn + xres[(size_t)m * D_MODEL + n];
    }
}

// ---------------------------------------------------------------------------
extern "C" void kernel_launch(void* const* d_in, const int* in_sizes, int n_in,
                              void* d_out, int out_size, void* d_ws, size_t ws_size,
                              hipStream_t stream)
{
  const float* x      = (const float*)d_in[0];
  const float* ln_g   = (const float*)d_in[1];
  const float* ln_b   = (const float*)d_in[2];
  const float* W_in   = (const float*)d_in[3];
  const float* b_in   = (const float*)d_in[4];
  const float* conv_k = (const float*)d_in[5];
  const float* conv_b = (const float*)d_in[6];
  const float* Wa     = (const float*)d_in[7];
  const float* ba     = (const float*)d_in[8];
  const float* Wb     = (const float*)d_in[9];
  const float* bb     = (const float*)d_in[10];
  const float* Wc     = (const float*)d_in[11];
  const float* bc     = (const float*)d_in[12];
  const float* Wd     = (const float*)d_in[13];
  const float* bd     = (const float*)d_in[14];
  const float* a0     = (const float*)d_in[15];
  const float* b0     = (const float*)d_in[16];
  const float* c0     = (const float*)d_in[17];
  const float* d0     = (const float*)d_in[18];
  const float* W_out  = (const float*)d_in[19];
  const float* b_out  = (const float*)d_in[20];
  float* out = (float*)d_out;

  // ----- workspace carve-up -----
  char* w = (char*)d_ws;
  size_t off = 0;
  auto take = [&](size_t bytes) -> void* {
    void* p = w + off;
    off = (off + bytes + 255) & ~(size_t)255;
    return p;
  };
  __bf16* Wt_in  = (__bf16*)take((size_t)2048 * 1024 * 2);
  __bf16* Wt_g   = (__bf16*)take((size_t)4096 * 1024 * 2);
  __bf16* Wt_out = (__bf16*)take((size_t)1024 * 1024 * 2);
  __bf16* xn     = (__bf16*)take((size_t)MTOT * D_MODEL * 2);
  float*  xz     = (float*) take((size_t)MTOT * 2 * D_MODEL * 4);
  __bf16* xconv  = (__bf16*)take((size_t)MTOT * D_MODEL * 2);
  __bf16* gates  = (__bf16*)take((size_t)MTOT * 4096 * 2);
  __bf16* ybuf   = (__bf16*)take((size_t)MTOT * D_MODEL * 2);
  float*  chP    = (float*) take((size_t)BATCH * NCHUNK * D_MODEL * 4);
  float*  chS    = (float*) take((size_t)BATCH * NCHUNK * D_MODEL * 4);
  float*  h0     = (float*) take((size_t)BATCH * NCHUNK * D_MODEL * 4);
  (void)ws_size; (void)n_in; (void)in_sizes; (void)out_size;

  // ----- weight convert/transpose (bf16, N x K) -----
  {
    int tot = 2048 * 1024;
    cvt_tr_kernel<<<(tot + 255) / 256, 256, 0, stream>>>(W_in, Wt_in, 1024, 2048);
    tot = 1024 * 1024;
    int blk = (tot + 255) / 256;
    cvt_tr_kernel<<<blk, 256, 0, stream>>>(Wa, Wt_g + (size_t)0 * 1024 * 1024, 1024, 1024);
    cvt_tr_kernel<<<blk, 256, 0, stream>>>(Wb, Wt_g + (size_t)1 * 1024 * 1024, 1024, 1024);
    cvt_tr_kernel<<<blk, 256, 0, stream>>>(Wc, Wt_g + (size_t)2 * 1024 * 1024, 1024, 1024);
    cvt_tr_kernel<<<blk, 256, 0, stream>>>(Wd, Wt_g + (size_t)3 * 1024 * 1024, 1024, 1024);
    cvt_tr_kernel<<<blk, 256, 0, stream>>>(W_out, Wt_out, 1024, 1024);
  }

  // ----- pipeline -----
  ln_kernel<<<MTOT, 256, 0, stream>>>(x, ln_g, ln_b, xn);

  gemm_in_kernel<<<dim3(MTOT / 64, 2048 / 64), 128, 0, stream>>>(xn, Wt_in, b_in, xz);

  conv_kernel<<<(MTOT * D_MODEL) / 256, 256, 0, stream>>>(xz, conv_k, conv_b, xconv);

  gemm_gates_kernel<<<dim3(MTOT / 64, 4096 / 64), 128, 0, stream>>>(
      xconv, Wt_g, ba, bb, bc, bd, a0, b0, c0, d0, gates);

  scan1_kernel<<<(BATCH * NCHUNK * D_MODEL) / 256, 256, 0, stream>>>(gates, xz, chP, chS);
  scan2_kernel<<<(BATCH * D_MODEL) / 256, 256, 0, stream>>>(chP, chS, h0);
  scan3_kernel<<<(BATCH * NCHUNK * D_MODEL) / 256, 256, 0, stream>>>(gates, xz, h0, ybuf);

  gemm_out_kernel<<<dim3(MTOT / 64, 1024 / 64), 128, 0, stream>>>(ybuf, Wt_out, b_out, x, out);
}
